// TransformerEncoderBlock_59261958750315
// MI455X (gfx1250) — compile-verified, matrix operand order checked by reading, not run on hardware
//
#include <hip/hip_runtime.h>

// ---------------------------------------------------------------------------
// Transformer encoder block for MI455X (gfx1250, wave32, WMMA).
// All GEMMs: C = act(alpha * A(bf16)[MxK] @ Bt(bf16)[NxK]^T + bias) + resid
// using v_wmma_f32_16x16x32_bf16 with double-buffered
// GLOBAL_LOAD_ASYNC_TO_LDS_B128 staging (ASYNCcnt-tracked DMA).
// ---------------------------------------------------------------------------

#define BATCH 2
#define SEQ   2048
#define EMB   1024
#define HEADS 16
#define HDIM  64
#define E3    3072   // 3*EMB
#define FF    4096   // EXP*EMB

typedef __attribute__((ext_vector_type(16))) __bf16 v16bf;
typedef __attribute__((ext_vector_type(8)))  __bf16 v8bf;
typedef __attribute__((ext_vector_type(8)))  float  v8f;

#define LDSS 40            // padded LDS row stride (elements) for 32-wide K tile

__device__ __forceinline__ unsigned short f32_to_bf16(float f) {
  unsigned u = __float_as_uint(f);
  u += 0x7FFFu + ((u >> 16) & 1u);          // round-to-nearest-even
  return (unsigned short)(u >> 16);
}

// Async DMA: 32 bytes global -> LDS, no VGPR staging. IOFFSET is added to both
// the LDS (VDST) and global (VADDR) addresses per ISA 08_async_tensor §4.4.
__device__ __forceinline__ void async_copy_32B(unsigned short* ldsDst,
                                               const unsigned short* gsrc) {
  // generic pointer to LDS: low 32 bits are the wave-relative LDS byte offset
  unsigned l = (unsigned)(size_t)ldsDst;
  unsigned long long g = (unsigned long long)(size_t)gsrc;
  asm volatile("global_load_async_to_lds_b128 %0, %1, off"
               :: "v"(l), "v"(g) : "memory");
  asm volatile("global_load_async_to_lds_b128 %0, %1, off offset:16"
               :: "v"(l), "v"(g) : "memory");
}

__device__ __forceinline__ void zero_32B(unsigned short* ldsDst) {
  uint4 z = make_uint4(0, 0, 0, 0);
  *(uint4*)(ldsDst)     = z;
  *(uint4*)(ldsDst + 8) = z;
}

__device__ __forceinline__ v16bf load_frag(const unsigned short* lds, int rowBase, int lane) {
  // 16-bit A/B fragment layout (ISA 7.12.2): lane l holds row (l&15),
  // K-halves selected by (l>>4): VGPR0..3 = K[kb..kb+7], VGPR4..7 = K[16+kb..16+kb+7]
  const int row = rowBase + (lane & 15);
  const int kb  = (lane >> 4) << 3;
  const __bf16* p = (const __bf16*)(lds + row * LDSS);
  v8bf lo = *(const v8bf*)(p + kb);
  v8bf hi = *(const v8bf*)(p + 16 + kb);
  return __builtin_shufflevector(lo, hi, 0, 1, 2, 3, 4, 5, 6, 7,
                                          8, 9, 10, 11, 12, 13, 14, 15);
}

// ---------------------------------------------------------------------------
// Tiled bf16 WMMA GEMM. Block = 256 threads = 8 waves; 128x128 tile per block,
// 32x64 per wave (2x4 subtiles of 16x16, K-step 32). Double-buffered async
// global->LDS staging overlaps DMA of tile k+1 with WMMA on tile k.
// ---------------------------------------------------------------------------
__global__ __launch_bounds__(256)
void gemm_bf16_wmma(const unsigned short* __restrict__ A,  int lda,
                    const unsigned short* __restrict__ Bt, int ldbt,
                    float*          __restrict__ Cf, int ldc,
                    unsigned short* __restrict__ Cb, int ldcb,
                    const float*    __restrict__ bias,
                    const float*    __restrict__ resid, int ldr,
                    int M, int N, int K, float alpha, int gelu)
{
  __shared__ unsigned short lds[2][2][128 * LDSS];   // [buf][A=0/B=1][tile]

  const int tid    = threadIdx.x;
  const int lane   = tid & 31;
  const int wave   = tid >> 5;
  const int blockM = blockIdx.y * 128;
  const int blockN = blockIdx.x * 128;
  const int waveM  = (wave & 3) * 32;   // 4 waves along M
  const int waveN  = (wave >> 2) * 64;  // 2 waves along N

  v8f acc[2][4];
#pragma unroll
  for (int mi = 0; mi < 2; ++mi)
#pragma unroll
    for (int ni = 0; ni < 4; ++ni)
#pragma unroll
      for (int r = 0; r < 8; ++r) acc[mi][ni][r] = 0.0f;

  const int srow = tid >> 1;          // staging row  0..127
  const int scol = (tid & 1) * 16;    // staging col half: 0 or 16

  auto stage = [&](int buf, int k0) {
    {
      const int gr = blockM + srow;
      unsigned short* dst = &lds[buf][0][srow * LDSS + scol];
      if (gr < M) async_copy_32B(dst, A + (size_t)gr * lda + k0 + scol);
      else        zero_32B(dst);
    }
    {
      const int gr = blockN + srow;
      unsigned short* dst = &lds[buf][1][srow * LDSS + scol];
      if (gr < N) async_copy_32B(dst, Bt + (size_t)gr * ldbt + k0 + scol);
      else        zero_32B(dst);
    }
  };

  // prologue: fill buffer 0
  stage(0, 0);
  asm volatile("s_wait_asynccnt 0x0" ::: "memory");
  __syncthreads();

  int buf = 0;
  for (int k0 = 0; k0 < K; k0 += 32) {
    // kick off DMA for next K-tile into the other buffer
    if (k0 + 32 < K) stage(buf ^ 1, k0 + 32);

    v16bf af[2], bfr[4];
#pragma unroll
    for (int mi = 0; mi < 2; ++mi) af[mi] = load_frag(&lds[buf][0][0], waveM + mi * 16, lane);
#pragma unroll
    for (int ni = 0; ni < 4; ++ni) bfr[ni] = load_frag(&lds[buf][1][0], waveN + ni * 16, lane);

#pragma unroll
    for (int mi = 0; mi < 2; ++mi)
#pragma unroll
      for (int ni = 0; ni < 4; ++ni)
        acc[mi][ni] = __builtin_amdgcn_wmma_f32_16x16x32_bf16(
            /*neg_a=*/false, af[mi], /*neg_b=*/false, bfr[ni],
            /*c_mod=*/(short)0, acc[mi][ni], /*reuse_a=*/false, /*reuse_b=*/false);

    // own async DMA complete, then block-wide barrier => everyone's complete
    asm volatile("s_wait_asynccnt 0x0" ::: "memory");
    __syncthreads();
    buf ^= 1;
  }

  // ---- epilogue: alpha, bias, gelu, residual, dual-format store ----
  // C/D layout (ISA 7.12.2): lanes 0-15 -> M rows 0..7, lanes 16-31 -> rows 8..15; N = lane&15.
#pragma unroll
  for (int mi = 0; mi < 2; ++mi) {
#pragma unroll
    for (int ni = 0; ni < 4; ++ni) {
      const int col = blockN + waveN + ni * 16 + (lane & 15);
      if (col >= N) continue;
      const float bi = bias ? bias[col] : 0.0f;
      const int mrow0 = blockM + waveM + mi * 16 + ((lane >> 4) << 3);
#pragma unroll
      for (int r = 0; r < 8; ++r) {
        const int row = mrow0 + r;
        if (row >= M) continue;
        float v = acc[mi][ni][r] * alpha + bi;
        if (gelu) v = 0.5f * v * (1.0f + erff(v * 0.70710678118654752f));
        if (resid) v += resid[(size_t)row * ldr + col];
        if (Cf) Cf[(size_t)row * ldc  + col] = v;
        if (Cb) Cb[(size_t)row * ldcb + col] = f32_to_bf16(v);
      }
    }
  }
}

// ---------------------------------------------------------------------------
// LayerNorm over E=1024, writes bf16 for GEMM consumption. One block per row.
// ---------------------------------------------------------------------------
__global__ __launch_bounds__(256)
void layernorm_to_bf16(const float* __restrict__ x, const float* __restrict__ g,
                       const float* __restrict__ b, unsigned short* __restrict__ out)
{
  const int row = blockIdx.x;
  const int tid = threadIdx.x;
  const float* xr = x + (size_t)row * EMB;

  float v[4], s = 0.0f, s2 = 0.0f;
#pragma unroll
  for (int i = 0; i < 4; ++i) {
    v[i] = xr[tid + i * 256];
    s  += v[i];
    s2 += v[i] * v[i];
  }
  __shared__ float r1[256], r2[256];
  r1[tid] = s; r2[tid] = s2;
  __syncthreads();
  for (int o = 128; o > 0; o >>= 1) {
    if (tid < o) { r1[tid] += r1[tid + o]; r2[tid] += r2[tid + o]; }
    __syncthreads();
  }
  const float mu  = r1[0] * (1.0f / EMB);
  const float var = r2[0] * (1.0f / EMB) - mu * mu;
  const float inv = rsqrtf(var + 1e-5f);
#pragma unroll
  for (int i = 0; i < 4; ++i) {
    const int c = tid + i * 256;
    out[(size_t)row * EMB + c] = f32_to_bf16((v[i] - mu) * inv * g[c] + b[c]);
  }
}

// ---------------------------------------------------------------------------
// Row softmax over 2048 cols (f32 in, bf16 out). One block per row.
// ---------------------------------------------------------------------------
__global__ __launch_bounds__(256)
void softmax_to_bf16(const float* __restrict__ S, unsigned short* __restrict__ P)
{
  const int row = blockIdx.x;
  const int tid = threadIdx.x;
  const float* sr = S + (size_t)row * SEQ;

  float v[8], m = -3.0e38f;
#pragma unroll
  for (int i = 0; i < 8; ++i) { v[i] = sr[tid + i * 256]; m = fmaxf(m, v[i]); }

  __shared__ float red[256];
  red[tid] = m;
  __syncthreads();
  for (int o = 128; o > 0; o >>= 1) {
    if (tid < o) red[tid] = fmaxf(red[tid], red[tid + o]);
    __syncthreads();
  }
  m = red[0];
  __syncthreads();

  float s = 0.0f;
#pragma unroll
  for (int i = 0; i < 8; ++i) { v[i] = __expf(v[i] - m); s += v[i]; }
  red[tid] = s;
  __syncthreads();
  for (int o = 128; o > 0; o >>= 1) {
    if (tid < o) red[tid] += red[tid + o];
    __syncthreads();
  }
  const float inv = 1.0f / red[0];
#pragma unroll
  for (int i = 0; i < 8; ++i)
    P[(size_t)row * SEQ + tid + i * 256] = f32_to_bf16(v[i] * inv);
}

// ---------------------------------------------------------------------------
// dst[C x R] (bf16) = transpose(src[R x C] (f32))   -- weight pre-transpose
// ---------------------------------------------------------------------------
__global__ __launch_bounds__(256)
void cast_transpose_f32_bf16(const float* __restrict__ src,
                             unsigned short* __restrict__ dst, int R, int C)
{
  const size_t idx = (size_t)blockIdx.x * 256 + threadIdx.x;
  if (idx >= (size_t)R * C) return;
  const int r = (int)(idx % R);
  const int c = (int)(idx / R);
  dst[idx] = f32_to_bf16(src[(size_t)r * C + c]);
}

// ---------------------------------------------------------------------------
// Vt[b,h,d,l] = qkv_bf16[b, l, 2E + h*64 + d]   (so P@V becomes A x Bt form)
// ---------------------------------------------------------------------------
__global__ __launch_bounds__(256)
void build_vt(const unsigned short* __restrict__ qkv, unsigned short* __restrict__ vt)
{
  const size_t idx = (size_t)blockIdx.x * 256 + threadIdx.x;
  if (idx >= (size_t)BATCH * HEADS * HDIM * SEQ) return;
  const int l  = (int)(idx % SEQ);
  const int d  = (int)((idx / SEQ) % HDIM);
  const int h  = (int)((idx / ((size_t)SEQ * HDIM)) % HEADS);
  const int b  = (int)(idx / ((size_t)SEQ * HDIM * HEADS));
  vt[idx] = qkv[((size_t)b * SEQ + l) * E3 + 2 * EMB + h * HDIM + d];
}

// ---------------------------------------------------------------------------
// Host orchestration
// ---------------------------------------------------------------------------
static inline void launch_gemm(const unsigned short* A, int lda,
                               const unsigned short* Bt, int ldbt,
                               float* Cf, int ldc, unsigned short* Cb, int ldcb,
                               const float* bias, const float* resid, int ldr,
                               int M, int N, int K, float alpha, int gelu,
                               hipStream_t stream)
{
  dim3 grid((N + 127) / 128, (M + 127) / 128);
  gemm_bf16_wmma<<<grid, 256, 0, stream>>>(A, lda, Bt, ldbt, Cf, ldc, Cb, ldcb,
                                           bias, resid, ldr, M, N, K, alpha, gelu);
}

extern "C" void kernel_launch(void* const* d_in, const int* in_sizes, int n_in,
                              void* d_out, int out_size, void* d_ws, size_t ws_size,
                              hipStream_t stream) {
  const float* x      = (const float*)d_in[0];
  const float* ln1_g  = (const float*)d_in[1];
  const float* ln1_b  = (const float*)d_in[2];
  const float* W_qkv  = (const float*)d_in[3];
  const float* W_proj = (const float*)d_in[4];
  const float* b_proj = (const float*)d_in[5];
  const float* ln2_g  = (const float*)d_in[6];
  const float* ln2_b  = (const float*)d_in[7];
  const float* W1     = (const float*)d_in[8];
  const float* b1     = (const float*)d_in[9];
  const float* W2     = (const float*)d_in[10];
  const float* b2     = (const float*)d_in[11];
  float* out = (float*)d_out;

  const int ROWS = BATCH * SEQ;   // 4096

  char* ws = (char*)d_ws;
  size_t off = 0;
  auto alloc = [&](size_t bytes) -> void* {
    void* p = ws + off;
    off = (off + bytes + 255) & ~(size_t)255;
    return p;
  };

  unsigned short* h_bf    = (unsigned short*)alloc((size_t)ROWS * EMB * 2);
  unsigned short* qkv_bf  = (unsigned short*)alloc((size_t)ROWS * E3 * 2);
  unsigned short* WqkvT   = (unsigned short*)alloc((size_t)E3 * EMB * 2);
  unsigned short* WprojT  = (unsigned short*)alloc((size_t)EMB * EMB * 2);
  unsigned short* W1T     = (unsigned short*)alloc((size_t)FF * EMB * 2);
  unsigned short* W2T     = (unsigned short*)alloc((size_t)EMB * FF * 2);
  unsigned short* Vt      = (unsigned short*)alloc((size_t)BATCH * HEADS * HDIM * SEQ * 2);
  float*          Sbuf    = (float*)alloc((size_t)SEQ * SEQ * 4);
  unsigned short* Pbuf    = (unsigned short*)alloc((size_t)SEQ * SEQ * 2);
  unsigned short* attn_bf = (unsigned short*)alloc((size_t)ROWS * EMB * 2);
  float*          x1      = (float*)alloc((size_t)ROWS * EMB * 4);
  unsigned short* h1_bf   = (unsigned short*)alloc((size_t)ROWS * FF * 2);

  auto tgrid = [](size_t n) { return dim3((unsigned)((n + 255) / 256)); };

  // ---- weight pre-transposes (f32 -> bf16, [K x N] -> [N x K]) ----
  cast_transpose_f32_bf16<<<tgrid((size_t)EMB * E3), 256, 0, stream>>>(W_qkv,  WqkvT,  EMB, E3);
  cast_transpose_f32_bf16<<<tgrid((size_t)EMB * EMB), 256, 0, stream>>>(W_proj, WprojT, EMB, EMB);
  cast_transpose_f32_bf16<<<tgrid((size_t)EMB * FF), 256, 0, stream>>>(W1,     W1T,    EMB, FF);
  cast_transpose_f32_bf16<<<tgrid((size_t)FF * EMB), 256, 0, stream>>>(W2,     W2T,    FF,  EMB);

  // ---- LN1 -> h (bf16) ----
  layernorm_to_bf16<<<ROWS, 256, 0, stream>>>(x, ln1_g, ln1_b, h_bf);

  // ---- qkv = h @ W_qkv  -> bf16 [ROWS x 3E] ----
  launch_gemm(h_bf, EMB, WqkvT, EMB, nullptr, 0, qkv_bf, E3,
              nullptr, nullptr, 0, ROWS, E3, EMB, 1.0f, 0, stream);

  // ---- V^T per head ----
  build_vt<<<tgrid((size_t)BATCH * HEADS * HDIM * SEQ), 256, 0, stream>>>(qkv_bf, Vt);

  // ---- attention per (b,h) ----
  const float scale = 1.0f / 32.0f;  // 1/sqrt(E), E=1024
  for (int b = 0; b < BATCH; ++b) {
    for (int h = 0; h < HEADS; ++h) {
      const unsigned short* Qp = qkv_bf + (size_t)b * SEQ * E3 + (size_t)h * HDIM;
      const unsigned short* Kp = Qp + EMB;
      // S = Q @ K^T / 32   [2048 x 2048]
      launch_gemm(Qp, E3, Kp, E3, Sbuf, SEQ, nullptr, 0,
                  nullptr, nullptr, 0, SEQ, SEQ, HDIM, scale, 0, stream);
      // P = softmax(S) (bf16)
      softmax_to_bf16<<<SEQ, 256, 0, stream>>>(Sbuf, Pbuf);
      // O = P @ V  -> attn_bf[b, :, h*64 : h*64+64]
      launch_gemm(Pbuf, SEQ, Vt + ((size_t)b * HEADS + h) * HDIM * SEQ, SEQ,
                  nullptr, 0, attn_bf + (size_t)b * SEQ * EMB + (size_t)h * HDIM, EMB,
                  nullptr, nullptr, 0, SEQ, HDIM, SEQ, 1.0f, 0, stream);
    }
  }

  // ---- x1 = x + attn @ W_proj + b_proj  (f32) ----
  launch_gemm(attn_bf, EMB, WprojT, EMB, x1, EMB, nullptr, 0,
              b_proj, x, EMB, ROWS, EMB, EMB, 1.0f, 0, stream);

  // ---- LN2 -> h (bf16) ----
  layernorm_to_bf16<<<ROWS, 256, 0, stream>>>(x1, ln2_g, ln2_b, h_bf);

  // ---- h1 = gelu(h @ W1 + b1)  (bf16) ----
  launch_gemm(h_bf, EMB, W1T, EMB, nullptr, 0, h1_bf, FF,
              b1, nullptr, 0, ROWS, FF, EMB, 1.0f, 1, stream);

  // ---- out = x1 + h1 @ W2 + b2  (f32) ----
  launch_gemm(h1_bf, FF, W2T, FF, out, EMB, nullptr, 0,
              b2, x1, EMB, ROWS, EMB, FF, 1.0f, 0, stream);
}
